// Lite_Temp_Causal_Address_63505386438896
// MI455X (gfx1250) — matrix operand order repaired and checked by reading, not compiled, not run on hardware
//
#include <hip/hip_runtime.h>
#include <hip/hip_bf16.h>

// Problem constants
constexpr int kB = 2;
constexpr int kT = 2048;
constexpr int kD = 512;
constexpr int kK = 64;
constexpr int kTP  = kT + 96;   // q16p time extent: 64 left zero-pad + 32 right zero-pad
constexpr int kTQT = kT + 64;   // qT row extent: 64 left zero-pad rows

typedef __attribute__((ext_vector_type(16))) _Float16 v16h;
typedef __attribute__((ext_vector_type(8)))  _Float16 v8h;
typedef __attribute__((ext_vector_type(8)))  float    v8f;

// ---------------------------------------------------------------------------
// Kernel 1: q = gelu(conv1d(x, w, pad=1) + b), written as fp16 in two layouts:
//   q16p[b][c][tp]  (tp = t + 64, time-contiguous, zero halo)   -> context A-frags
//   qT  [b][tr][c]  (tr = t + 64, channel-contiguous, zero halo) -> gram A/B-frags
// ---------------------------------------------------------------------------
__global__ void prep_kernel(const float* __restrict__ x,
                            const float* __restrict__ w,
                            const float* __restrict__ bias,
                            _Float16* __restrict__ q16p,
                            _Float16* __restrict__ qT) {
  int idx = blockIdx.x * blockDim.x + threadIdx.x;
  if (idx >= kB * kD * kT) return;
  int t = idx % kT;
  int c = (idx / kT) % kD;
  int b = idx / (kT * kD);

  const float* xb = x + (size_t)b * kT;
  float xm = (t > 0)      ? xb[t - 1] : 0.0f;
  float x0 = xb[t];
  float xp = (t < kT - 1) ? xb[t + 1] : 0.0f;
  float v = xm * w[c * 3 + 0] + x0 * w[c * 3 + 1] + xp * w[c * 3 + 2] + bias[c];
  float g = 0.5f * v * (1.0f + erff(v * 0.70710678118654752f));  // exact gelu
  _Float16 h = (_Float16)g;

  q16p[((size_t)b * kD + c) * kTP + (t + 64)] = h;
  qT[((size_t)b * kTQT + (t + 64)) * (size_t)kD + c] = h;
}

// ---------------------------------------------------------------------------
// Kernel 2: one wave (32 lanes) per 16-timestep tile.
//  Phase 1: 5 banded Gram blocks G_ij = Q_i^T Q_j via v_wmma_f32_16x16x32_f16
//  Phase 2: softmax over k in LDS, attn staged transposed (fp16) in LDS
//  Phase 3: context tiles = q_tile * attn^T via WMMA, scaled by tanh(gate)
// ---------------------------------------------------------------------------
__global__ void __launch_bounds__(32)
attn_kernel(const _Float16* __restrict__ q16p,
            const _Float16* __restrict__ qT,
            const float* __restrict__ gate,
            float* __restrict__ out) {
  const int lane = threadIdx.x;       // 0..31
  const int tile = blockIdx.x;        // 0 .. kB*kT/16 - 1
  const int b   = tile / (kT / 16);
  const int t0  = (tile % (kT / 16)) * 16;
  const int n   = lane & 15;
  const int grp = lane >> 4;

  const _Float16* qTb = qT + (size_t)b * kTQT * kD;

  // ----- Phase 1: energy Gram band -----
  v8f acc[5];
#pragma unroll
  for (int jj = 0; jj < 5; ++jj) acc[jj] = (v8f){0.f,0.f,0.f,0.f,0.f,0.f,0.f,0.f};

  {
    const _Float16* aBase = qTb + (size_t)(t0 + 64) * kD;  // row t0 (padded coords)
    for (int c0 = 0; c0 < kD; c0 += 32) {
      // A fragment: A[m][k] = q[t0+m][c0+k]; per-lane two contiguous 16B chunks
      const _Float16* pa = aBase + (size_t)n * kD + c0 + grp * 16;
      v8h alo = *(const v8h*)(pa);
      v8h ahi = *(const v8h*)(pa + 8);
      v16h a;
#pragma unroll
      for (int i = 0; i < 8; ++i) { a[i] = alo[i]; a[8 + i] = ahi[i]; }

#pragma unroll
      for (int jj = 0; jj < 5; ++jj) {
        // B fragment: B[k][nn] = q[u0+nn][c0+k], u0 = t0-64+16*jj (padded: +64)
        const _Float16* pb = qTb + (size_t)(t0 + 16 * jj + n) * kD + c0 + grp * 16;
        v8h blo = *(const v8h*)(pb);
        v8h bhi = *(const v8h*)(pb + 8);
        v16h bm;
#pragma unroll
        for (int i = 0; i < 8; ++i) { bm[i] = blo[i]; bm[8 + i] = bhi[i]; }
        acc[jj] = __builtin_amdgcn_wmma_f32_16x16x32_f16(
            false, a, false, bm, (short)0, acc[jj], false, false);
      }
    }
  }

  // ----- Phase 2: scatter energy -> LDS, softmax, stage attn^T (fp16) -----
  __shared__ float e[16][68];                      // padded stride vs 64 banks
  __shared__ __align__(16) _Float16 PT[16][96];    // PT[t_col][u_rel], u_rel = u-(t0-64)

  const float scale = 0.04419417382415922f;        // 1/sqrt(512)
#pragma unroll
  for (int jj = 0; jj < 5; ++jj) {
#pragma unroll
    for (int r = 0; r < 8; ++r) {
      int m = r + grp * 8;                         // C-frag row
      int k = 16 * jj + n - m - 1;                 // k = u - t + 63
      if (k >= 0 && k < kK) e[m][k] = acc[jj][r] * scale;
    }
  }
  {   // zero the attn staging buffer (covers masked u positions)
    unsigned int* pz = (unsigned int*)&PT[0][0];
    for (int i = lane; i < 16 * 96 / 2; i += 32) pz[i] = 0u;
  }
  __syncthreads();

  if (lane < 16) {
    const int m = lane;
    float mx = -3.0e38f;
    for (int k = 0; k < kK; ++k) mx = fmaxf(mx, e[m][k]);
    float s = 0.0f;
    for (int k = 0; k < kK; ++k) s += __expf(e[m][k] - mx);
    float inv = 1.0f / s;
    for (int k = 0; k < kK; ++k)
      PT[m][m + 1 + k] = (_Float16)(__expf(e[m][k] - mx) * inv);  // u_rel = m+1+k
  }
  __syncthreads();

  // ----- Phase 3: context = q * attn^T (banded, 96-wide u window) -----
  v16h bf[3];
#pragma unroll
  for (int uc = 0; uc < 3; ++uc) {   // B[k][nn] = PT[nn][uc*32 + k] (ds_load_b128 x2)
    const _Float16* p = &PT[n][uc * 32 + grp * 16];
    v8h lo = *(const v8h*)p;
    v8h hi = *(const v8h*)(p + 8);
#pragma unroll
    for (int i = 0; i < 8; ++i) { bf[uc][i] = lo[i]; bf[uc][8 + i] = hi[i]; }
  }

  const float tg = tanhf(gate[0]);
  const _Float16* qpb = q16p + (size_t)b * kD * kTP;
  float* outb = out + (size_t)b * kD * kT;

  for (int c0 = 0; c0 < kD; c0 += 16) {
    v8f cacc = (v8f){0.f,0.f,0.f,0.f,0.f,0.f,0.f,0.f};
#pragma unroll
    for (int uc = 0; uc < 3; ++uc) {
      // A[m][k] = q16p[c0+m][tp0 + k], tp0 = t0 + uc*32 (padded coords)
      const _Float16* pa = qpb + (size_t)(c0 + n) * kTP + t0 + uc * 32 + grp * 16;
      v8h alo = *(const v8h*)pa;
      v8h ahi = *(const v8h*)(pa + 8);
      v16h a;
#pragma unroll
      for (int i = 0; i < 8; ++i) { a[i] = alo[i]; a[8 + i] = ahi[i]; }
      cacc = __builtin_amdgcn_wmma_f32_16x16x32_f16(
          false, a, false, bf[uc], (short)0, cacc, false, false);
    }
#pragma unroll
    for (int r = 0; r < 8; ++r) {
      int c = c0 + grp * 8 + r;                    // C-frag: m = r + grp*8
      outb[(size_t)c * kT + t0 + n] = cacc[r] * tg;
    }
  }
}

// ---------------------------------------------------------------------------
// Launch
// ---------------------------------------------------------------------------
extern "C" void kernel_launch(void* const* d_in, const int* in_sizes, int n_in,
                              void* d_out, int out_size, void* d_ws, size_t ws_size,
                              hipStream_t stream) {
  const float* x      = (const float*)d_in[0];
  const float* conv_w = (const float*)d_in[1];
  const float* conv_b = (const float*)d_in[2];
  const float* gate   = (const float*)d_in[3];
  float* out = (float*)d_out;

  constexpr size_t q16p_bytes = (size_t)kB * kD * kTP * sizeof(_Float16);
  constexpr size_t qT_off     = (q16p_bytes + 255) & ~(size_t)255;
  constexpr size_t qT_bytes   = (size_t)kB * kTQT * kD * sizeof(_Float16);
  constexpr size_t total      = qT_off + qT_bytes;   // ~8.7 MB

  _Float16* q16p = (_Float16*)d_ws;
  _Float16* qTw  = (_Float16*)((char*)d_ws + qT_off);

  // Zero halos (and everything else) each call; capturable as a memset node.
  hipMemsetAsync(d_ws, 0, total, stream);

  int nPrep = kB * kD * kT;
  prep_kernel<<<(nPrep + 255) / 256, 256, 0, stream>>>(x, conv_w, conv_b, q16p, qTw);

  int nTiles = kB * (kT / 16);   // 256 tiles, one wave32 each
  attn_kernel<<<nTiles, 32, 0, stream>>>(q16p, qTw, gate, out);
}
